// LSTM_48850958024796
// MI455X (gfx1250) — matrix operand order repaired and checked by reading, not compiled, not run on hardware
//
#include <hip/hip_runtime.h>
#include <hip/hip_bf16.h>

// ---------------------------------------------------------------------------
// Bidirectional 2-layer LSTM for MI455X (gfx1250), bf16 WMMA.
//   B=1024, T=128, IN=5, H=100. Gate layout padded: n = gate*128 + j (NGP=512)
//   so each wave owns one 16-wide j-tile and accumulates all 4 gates for the
//   same columns -> cell update is done fully in registers (no gate LDS).
//   Combined weights Wc[n][KC], KC = 128 (hidden, padded) + KI (input padded):
//   KI = 32 (layer 0) or 256 (layer 1). Hidden-part B fragments are register-
//   resident across all 128 timesteps; A fragments batch-loaded per step so
//   the WMMA burst runs without interleaved LDS waits. Branch-free fast
//   sigmoid/tanh (exp + fast divide). Workspace ~77 MB.
// ---------------------------------------------------------------------------

typedef __attribute__((ext_vector_type(16))) __bf16 v16bf;
typedef __attribute__((ext_vector_type(8)))  __bf16 v8bf;
typedef __attribute__((ext_vector_type(8)))  float  v8f;

#define BSZ 1024
#define TSZ 128
#define HH  100
#define HP  128          // per-gate column stride (padded H)
#define NGP 512          // 4 * HP padded gate rows
#define KHP 128          // padded hidden K

// Branch-free activations (no libm range-check ladders, no exec divergence).
// tanh(x) = 1 - 2/(exp(2x)+1): exp->inf gives +1, exp->0 gives -1; no clamp.
__device__ __forceinline__ float sigm(float x) {
    return __fdividef(1.f, 1.f + __expf(-x));
}
__device__ __forceinline__ float ftanh(float x) {
    return 1.f - __fdividef(2.f, __expf(2.f * x) + 1.f);
}
__device__ __forceinline__ v16bf cat8(v8bf lo, v8bf hi) {
    return __builtin_shufflevector(lo, hi, 0,1,2,3,4,5,6,7,8,9,10,11,12,13,14,15);
}

// ---------------------------------------------------------------------------
// Prepack kernels
// ---------------------------------------------------------------------------
__global__ void prep_x_kernel(const float* __restrict__ x, __bf16* __restrict__ xb, int total) {
    int i = blockIdx.x * 256 + threadIdx.x;
    if (i >= total) return;
    int bt = i >> 5, k = i & 31;            // [B*T][32]
    float v = (k < 5) ? x[bt * 5 + k] : 0.f;
    xb[i] = (__bf16)v;
}

__global__ void zero16_kernel(uint4* __restrict__ p, int n16) {
    int i = blockIdx.x * 256 + threadIdx.x;
    if (i < n16) p[i] = make_uint4(0u, 0u, 0u, 0u);
}

// Wc[np][k], np = g*128 + j: j<100 -> gate row g*100+j else zeros.
//   k<128: w_hh[row][k] (k<100 else 0);  k>=128: w_ih[row][k-128] (<din else 0)
__global__ void prep_wc_kernel(const float* __restrict__ whh, const float* __restrict__ wih,
                               int din, int KC, __bf16* __restrict__ wc, int total) {
    int i = blockIdx.x * 256 + threadIdx.x;
    if (i >= total) return;
    int np = i / KC, k = i % KC;
    int g = np >> 7, j = np & 127;
    float v = 0.f;
    if (j < HH) {
        int row = g * HH + j;
        if (k < KHP) { if (k < HH) v = whh[row * HH + k]; }
        else         { int ki = k - KHP; if (ki < din) v = wih[row * din + ki]; }
    }
    wc[i] = (__bf16)v;
}

__global__ void prep_bsum_kernel(const float* __restrict__ bi, const float* __restrict__ bh,
                                 float* __restrict__ bs) {
    int i = blockIdx.x * 256 + threadIdx.x;
    if (i < NGP) {
        int g = i >> 7, j = i & 127;
        bs[i] = (j < HH) ? (bi[g * HH + j] + bh[g * HH + j]) : 0.f;
    }
}

// ---------------------------------------------------------------------------
// Persistent bidirectional LSTM layer. grid = (B/16, 2), blockIdx.y = dir.
// Each wave = one 16-wide j-tile, 4 gate accumulators in registers.
// ---------------------------------------------------------------------------
template <int KI>
__global__ __launch_bounds__(256, 1) void lstm_layer_kernel(
    const __bf16* __restrict__ wcf, const __bf16* __restrict__ wcb,
    const float* __restrict__ bsf, const float* __restrict__ bsb,
    const __bf16* __restrict__ in_seq,
    __bf16* __restrict__ out_seq, float* __restrict__ out_last)
{
    constexpr int KC  = KHP + KI;
    constexpr int NKH = KHP / 32;        // hidden K chunks (4)
    constexpr int NKI = KI / 32;         // input K chunks (1 or 8)
    constexpr int NKT = NKH + NKI;

    const int dir  = blockIdx.y;
    const __bf16* wc   = dir ? wcb : wcf;
    const float*  bsum = dir ? bsb : bsf;
    const int out_off = dir * HH;
    const int b0   = blockIdx.x * 16;
    const int tid  = threadIdx.x;
    const int lane = tid & 31;
    const int wave = tid >> 5;
    const int half = lane >> 4;          // 0: lanes 0-15, 1: lanes 16-31
    const int mrow = lane & 15;
    const int j    = wave * 16 + mrow;   // this lane's hidden column (0..127)

    __shared__ __align__(32) __bf16 Hs[16 * KHP];   // hidden state (bf16, padded)
    __shared__ __align__(32) __bf16 Is[16 * KI];    // staged input tile

    for (int i = tid; i < 16 * KHP; i += 256) Hs[i] = (__bf16)0.f;

    // ---- time-invariant B fragments: hidden part register-resident
    v16bf bh[4][NKH];
#pragma unroll
    for (int g = 0; g < 4; ++g) {
        const __bf16* wrow = wc + (size_t)(g * HP + j) * KC + half * 16;
#pragma unroll
        for (int kc = 0; kc < NKH; ++kc)
            bh[g][kc] = *(const v16bf*)(wrow + kc * 32);
    }
    v16bf bic[4];                         // input part, cached only for KI==32
    if constexpr (KI == 32) {
#pragma unroll
        for (int g = 0; g < 4; ++g)
            bic[g] = *(const v16bf*)(wc + (size_t)(g * HP + j) * KC + KHP + half * 16);
    }

    float bsl[4];
#pragma unroll
    for (int g = 0; g < 4; ++g) bsl[g] = bsum[g * HP + j];

    float creg[8];
#pragma unroll
    for (int v = 0; v < 8; ++v) creg[v] = 0.f;

    constexpr int IS_VECS = 16 * KI / 8;  // uint4 chunks in input tile

    for (int s = 0; s < TSZ; ++s) {
        const int t = dir ? (TSZ - 1 - s) : s;

        // ---- stage input tile cooperatively; prefetch next step's rows
        for (int idx = tid; idx < IS_VECS; idx += 256) {
            const int r = idx / (KI / 8), c8 = idx % (KI / 8);
            const __bf16* src = in_seq + ((size_t)(b0 + r) * TSZ + t) * KI + c8 * 8;
            *(uint4*)&Is[idx * 8] = *(const uint4*)src;
            if (s + 1 < TSZ) {
                const int tn = dir ? (t - 1) : (t + 1);
                __builtin_prefetch(in_seq + ((size_t)(b0 + r) * TSZ + tn) * KI + c8 * 8, 0, 1);
            }
        }
        __syncthreads();   // Is ready; previous step's Hs writes visible

        // ---- batch-load ALL A fragments first (one LDS burst, one wait)
        v16bf af[NKT];
        const __bf16* hrow = &Hs[mrow * KHP + half * 16];
        const __bf16* irow = &Is[mrow * KI + half * 16];
#pragma unroll
        for (int kc = 0; kc < NKH; ++kc)
            af[kc] = cat8(*(const v8bf*)(hrow + kc * 32),
                          *(const v8bf*)(hrow + kc * 32 + 8));
#pragma unroll
        for (int kc = 0; kc < NKI; ++kc)
            af[NKH + kc] = cat8(*(const v8bf*)(irow + kc * 32),
                                *(const v8bf*)(irow + kc * 32 + 8));

        // ---- WMMA burst: acc[g] += [Hs | Is] x Wc^T
        v8f acc[4] = {};
#pragma unroll
        for (int kc = 0; kc < NKH; ++kc) {
#pragma unroll
            for (int g = 0; g < 4; ++g)
                acc[g] = __builtin_amdgcn_wmma_f32_16x16x32_bf16(
                    false, af[kc], false, bh[g][kc], (short)0, acc[g], false, false);
        }
#pragma unroll
        for (int kc = 0; kc < NKI; ++kc) {
#pragma unroll
            for (int g = 0; g < 4; ++g) {
                v16bf bf;
                if constexpr (KI == 32) bf = bic[g];
                else bf = *(const v16bf*)(wc + (size_t)(g * HP + j) * KC + KHP + kc * 32 + half * 16);
                acc[g] = __builtin_amdgcn_wmma_f32_16x16x32_bf16(
                    false, af[NKH + kc], false, bf, (short)0, acc[g], false, false);
            }
        }
        __syncthreads();   // all Hs reads done before the update below

        // ---- cell update fully in registers; lane owns column j, rows m=v+half*8
        if (j < HH) {
#pragma unroll
            for (int v = 0; v < 8; ++v) {
                const int m = v + half * 8;
                float gi = acc[0][v] + bsl[0];
                float gf = acc[1][v] + bsl[1];
                float gg = acc[2][v] + bsl[2];
                float go = acc[3][v] + bsl[3];
                float c = sigm(gf) * creg[v] + sigm(gi) * ftanh(gg);
                float h = sigm(go) * ftanh(c);
                creg[v] = c;
                Hs[m * KHP + j] = (__bf16)h;
                if (out_seq)
                    out_seq[((size_t)(b0 + m) * TSZ + t) * 256 + out_off + j] = (__bf16)h;
                if (out_last && t == TSZ - 1)
                    out_last[(b0 + m) * (2 * HH) + out_off + j] = h;
            }
        }
        // next iteration's first barrier orders Hs writes vs. matmul reads
    }
}

// ---------------------------------------------------------------------------
// FC (200 -> 3) + softmax
// ---------------------------------------------------------------------------
__global__ void fc_softmax_kernel(const float* __restrict__ hl,
                                  const float* __restrict__ fcw,
                                  const float* __restrict__ fcb,
                                  float* __restrict__ out) {
    int b = blockIdx.x * 256 + threadIdx.x;
    if (b >= BSZ) return;
    float l[3];
#pragma unroll
    for (int c = 0; c < 3; ++c) {
        float acc = fcb[c];
        for (int k = 0; k < 2 * HH; ++k)
            acc += hl[b * (2 * HH) + k] * fcw[c * (2 * HH) + k];
        l[c] = acc;
    }
    float mx = fmaxf(l[0], fmaxf(l[1], l[2]));
    float e0 = __expf(l[0] - mx), e1 = __expf(l[1] - mx), e2 = __expf(l[2] - mx);
    float inv = __fdividef(1.f, e0 + e1 + e2);
    out[b * 3 + 0] = e0 * inv;
    out[b * 3 + 1] = e1 * inv;
    out[b * 3 + 2] = e2 * inv;
}

// ---------------------------------------------------------------------------
// Host launcher
// ---------------------------------------------------------------------------
extern "C" void kernel_launch(void* const* d_in, const int* in_sizes, int n_in,
                              void* d_out, int out_size, void* d_ws, size_t ws_size,
                              hipStream_t stream) {
    (void)in_sizes; (void)n_in; (void)out_size; (void)ws_size;

    const float* x        = (const float*)d_in[0];
    const float* w_ih_l0f = (const float*)d_in[1];
    const float* w_hh_l0f = (const float*)d_in[2];
    const float* b_ih_l0f = (const float*)d_in[3];
    const float* b_hh_l0f = (const float*)d_in[4];
    const float* w_ih_l0b = (const float*)d_in[5];
    const float* w_hh_l0b = (const float*)d_in[6];
    const float* b_ih_l0b = (const float*)d_in[7];
    const float* b_hh_l0b = (const float*)d_in[8];
    const float* w_ih_l1f = (const float*)d_in[9];
    const float* w_hh_l1f = (const float*)d_in[10];
    const float* b_ih_l1f = (const float*)d_in[11];
    const float* b_hh_l1f = (const float*)d_in[12];
    const float* w_ih_l1b = (const float*)d_in[13];
    const float* w_hh_l1b = (const float*)d_in[14];
    const float* b_ih_l1b = (const float*)d_in[15];
    const float* b_hh_l1b = (const float*)d_in[16];
    const float* fc_w     = (const float*)d_in[17];
    const float* fc_b     = (const float*)d_in[18];

    // ---- carve workspace (~77 MB)
    char* ws = (char*)d_ws;
    size_t off = 0;
    auto take = [&](size_t bytes) -> char* {
        char* p = ws + off;
        off += (bytes + 255) & ~(size_t)255;
        return p;
    };
    __bf16* x0bf = (__bf16*)take((size_t)BSZ * TSZ * 32 * 2);    //  8 MB
    __bf16* h0bf = (__bf16*)take((size_t)BSZ * TSZ * 256 * 2);   // 67 MB
    __bf16* wc0f = (__bf16*)take((size_t)NGP * 160 * 2);
    __bf16* wc0b = (__bf16*)take((size_t)NGP * 160 * 2);
    __bf16* wc1f = (__bf16*)take((size_t)NGP * 384 * 2);
    __bf16* wc1b = (__bf16*)take((size_t)NGP * 384 * 2);
    float* bs0f = (float*)take(NGP * 4);
    float* bs0b = (float*)take(NGP * 4);
    float* bs1f = (float*)take(NGP * 4);
    float* bs1b = (float*)take(NGP * 4);
    float* h1last = (float*)take((size_t)BSZ * 2 * HH * 4);      // 0.8 MB

    // ---- prepack
    prep_x_kernel<<<(BSZ * TSZ * 32) / 256, 256, 0, stream>>>(x, x0bf, BSZ * TSZ * 32);
    zero16_kernel<<<((size_t)BSZ * TSZ * 256 * 2 / 16 + 255) / 256, 256, 0, stream>>>(
        (uint4*)h0bf, (int)((size_t)BSZ * TSZ * 256 * 2 / 16));
    prep_wc_kernel<<<(NGP * 160 + 255) / 256, 256, 0, stream>>>(w_hh_l0f, w_ih_l0f, 5,   160, wc0f, NGP * 160);
    prep_wc_kernel<<<(NGP * 160 + 255) / 256, 256, 0, stream>>>(w_hh_l0b, w_ih_l0b, 5,   160, wc0b, NGP * 160);
    prep_wc_kernel<<<(NGP * 384 + 255) / 256, 256, 0, stream>>>(w_hh_l1f, w_ih_l1f, 200, 384, wc1f, NGP * 384);
    prep_wc_kernel<<<(NGP * 384 + 255) / 256, 256, 0, stream>>>(w_hh_l1b, w_ih_l1b, 200, 384, wc1b, NGP * 384);
    prep_bsum_kernel<<<2, 256, 0, stream>>>(b_ih_l0f, b_hh_l0f, bs0f);
    prep_bsum_kernel<<<2, 256, 0, stream>>>(b_ih_l0b, b_hh_l0b, bs0b);
    prep_bsum_kernel<<<2, 256, 0, stream>>>(b_ih_l1f, b_hh_l1f, bs1f);
    prep_bsum_kernel<<<2, 256, 0, stream>>>(b_ih_l1b, b_hh_l1b, bs1b);

    // ---- layer 0 (both directions concurrently) -> h0 sequence (bf16, padded 256)
    lstm_layer_kernel<32><<<dim3(BSZ / 16, 2), 256, 0, stream>>>(
        wc0f, wc0b, bs0f, bs0b, x0bf, h0bf, nullptr);
    // ---- layer 1: only t = T-1 hidden needed downstream
    lstm_layer_kernel<256><<<dim3(BSZ / 16, 2), 256, 0, stream>>>(
        wc1f, wc1b, bs1f, bs1b, h0bf, nullptr, h1last);

    // ---- FC + softmax
    fc_softmax_kernel<<<(BSZ + 255) / 256, 256, 0, stream>>>(h1last, fc_w, fc_b, (float*)d_out);
}
